// Conv2D_17394617548934
// MI455X (gfx1250) — matrix operand order repaired and checked by reading, not compiled, not run on hardware
//
#include <hip/hip_runtime.h>

typedef __attribute__((ext_vector_type(16))) _Float16 v16h;
typedef __attribute__((ext_vector_type(8)))  _Float16 v8h;
typedef __attribute__((ext_vector_type(2)))  _Float16 v2h;
typedef __attribute__((ext_vector_type(8)))  float    v8f;
typedef __attribute__((ext_vector_type(4)))  float    v4f;
typedef int vi4 __attribute__((vector_size(16)));   // matches async-LDS builtin param type

#define B_      16
#define CIN     128
#define COUT    128
#define H_      224
#define W_      224
#define HW      (H_ * W_)
#define MT      112            // pixels per block (half a row); 112 = 7*16
#define COLS    (MT + 2)       // 114 input columns (halo of 1 each side)
#define CPAD    136            // padded channel stride (halves): 272B rows -> 16B aligned, low-conflict b128
#define NTILES  8              // 128 couts / 16
#define THREADS 224            // 7 waves of 32; wave w owns M-subtile w

#if __has_builtin(__builtin_amdgcn_global_load_async_to_lds_b128) && \
    __has_builtin(__builtin_amdgcn_s_wait_asynccnt)
#define HAVE_ASYNC_LDS 1
#else
#define HAVE_ASYNC_LDS 0
#endif

#if __has_builtin(__builtin_amdgcn_sched_barrier)
#define SCHED_FENCE() __builtin_amdgcn_sched_barrier(0)
#else
#define SCHED_FENCE() ((void)0)
#endif

// ---------------------------------------------------------------------------
// Prep: weights OIHW f32  ->  wt[tap][cout][cin] f16 in workspace.
// ---------------------------------------------------------------------------
__global__ void prep_weights_f16(const float* __restrict__ w, _Float16* __restrict__ wt) {
    int idx = blockIdx.x * 256 + threadIdx.x;
    if (idx >= 9 * COUT * CIN) return;
    int c = idx & (CIN - 1);
    int n = (idx >> 7) & (COUT - 1);
    int t = idx >> 14;                 // 128*128 = 2^14
    int ky = t / 3, kx = t - ky * 3;
    wt[idx] = (_Float16)w[((n * CIN + c) * 3 + ky) * 3 + kx];
}

// ---------------------------------------------------------------------------
// Main: implicit-GEMM conv via v_wmma_f32_16x16x32_f16.
// Per K-chunk: gather A + all 8 B fragments, sched-fence, then 8-WMMA burst.
// ---------------------------------------------------------------------------
__global__ __launch_bounds__(THREADS)
void conv3x3_wmma(const float* __restrict__ x, const _Float16* __restrict__ wt,
                  const float* __restrict__ bias, float* __restrict__ out) {
    extern __shared__ _Float16 smem[];
    _Float16* xs = smem;                        // [3][COLS][CPAD]  input patch, f16
    _Float16* ws = smem + 3 * COLS * CPAD;      // [COUT][CPAD]     one tap of weights

    const int tid  = threadIdx.x;
    const int lane = tid & 31;
    const int wave = tid >> 5;                  // 0..6  == M-subtile
    const int hi16 = lane >> 4;                 // lane group 0/1
    const int l16  = lane & 15;

    int bid = blockIdx.x;
    const int w0 = (bid & 1) * MT;
    bid >>= 1;
    const int h  = bid % H_;
    const int bb = bid / H_;

    // ---- stage input patch into LDS (f32 -> f16), lane-coalesced along w ----
    const float* xb = x + (size_t)bb * CIN * HW;
    for (int s = tid; s < 3 * COLS; s += THREADS) {       // 342 (ry,col) slots
        const int ry  = s / COLS;
        const int col = s - ry * COLS;
        const int ih  = h + ry - 1;
        const int iw  = w0 + col - 1;
        _Float16* dst = xs + (size_t)s * CPAD;
        if ((unsigned)ih < (unsigned)H_ && (unsigned)iw < (unsigned)W_) {
            const float* src = xb + (size_t)ih * W_ + iw;
            #pragma unroll 8
            for (int c = 0; c < CIN; c += 2) {
                v2h p = { (_Float16)src[(size_t)c * HW],
                          (_Float16)src[(size_t)(c + 1) * HW] };
                *(v2h*)(dst + c) = p;                     // ds_store_b32
            }
        } else {
            #pragma unroll 8
            for (int c = 0; c < CIN; c += 2) {
                v2h z = { (_Float16)0.f, (_Float16)0.f };
                *(v2h*)(dst + c) = z;
            }
        }
    }

    v8f acc[NTILES] = {};

    for (int t = 0; t < 9; ++t) {
        __syncthreads();                        // previous tap's ws fully consumed
        const _Float16* wsrc = wt + t * (COUT * CIN);
        // stage one tap of weights (32 KB) as 16B chunks: chunk idx -> row idx>>4, col (idx&15)*8
        for (int idx = tid; idx < (COUT * CIN) / 8; idx += THREADS) {
#if HAVE_ASYNC_LDS
            __builtin_amdgcn_global_load_async_to_lds_b128(
                (__attribute__((address_space(1))) vi4*)(wsrc + idx * 8),
                (__attribute__((address_space(3))) vi4*)(ws + (idx >> 4) * CPAD + (idx & 15) * 8),
                0, 0);
#else
            uint4 v = ((const uint4*)wsrc)[idx];
            *(uint4*)(ws + (idx >> 4) * CPAD + (idx & 15) * 8) = v;
#endif
        }
        if (t < 8)                              // warm L2 for next tap (global_prefetch_b8)
            __builtin_prefetch(wt + (t + 1) * (COUT * CIN) + tid * 64, 0, 1);
#if HAVE_ASYNC_LDS
        __builtin_amdgcn_s_wait_asynccnt(0);
#endif
        __syncthreads();

        const int ky = t / 3;
        const int kx = t - ky * 3;
        const _Float16* arow = xs + ((size_t)(ky * COLS) + (wave * 16 + l16 + kx)) * CPAD + hi16 * 8;
        const _Float16* brow = ws + (size_t)l16 * CPAD + hi16 * 8;

        // A 16x32 f16: lane(m=l16,grp hi16) holds local K {8g+j} and {16+8g+j}
        auto loadA = [&](int kc) -> v16h {
            const _Float16* ap = arow + kc * 32;
            v8h lo = *(const v8h*)(ap);
            v8h hi = *(const v8h*)(ap + 16);
            return __builtin_shufflevector(lo, hi, 0,1,2,3,4,5,6,7, 8,9,10,11,12,13,14,15);
        };
        // B 32x16 f16: lane(n=l16,grp hi16) holds local K {8g+j} and {16+8g+j}
        auto loadB = [&](int kc, int nt) -> v16h {
            const _Float16* bp = brow + nt * (16 * CPAD) + kc * 32;
            v8h lo = *(const v8h*)(bp);
            v8h hi = *(const v8h*)(bp + 16);
            return __builtin_shufflevector(lo, hi, 0,1,2,3,4,5,6,7, 8,9,10,11,12,13,14,15);
        };

        #pragma unroll
        for (int kc = 0; kc < 4; ++kc) {
            // gather phase: 18 independent ds_load_b128 into distinct registers
            v16h a = loadA(kc);
            v16h bf[NTILES];
            #pragma unroll
            for (int nt = 0; nt < NTILES; ++nt) bf[nt] = loadB(kc, nt);
            // hard scheduling fence: loads may not sink past this point, so the
            // wait inserter can emit graduated s_wait_dscnt (in-order DS returns)
            SCHED_FENCE();
            // burst phase: 8 back-to-back WMMAs; next kc's loads issue into the
            // shadow of this burst
            #pragma unroll
            for (int nt = 0; nt < NTILES; ++nt)
                acc[nt] = __builtin_amdgcn_wmma_f32_16x16x32_f16(
                    false, a, false, bf[nt], (short)0, acc[nt], false, false);
        }
    }

    // ---- store: D[m][n], m = 8*hi16 + r, n = l16; two aligned float4 per tile ----
    const int wbase = w0 + wave * 16 + hi16 * 8;
    #pragma unroll
    for (int nt = 0; nt < NTILES; ++nt) {
        const int co = nt * 16 + l16;
        const float bv = bias[co];
        float* dst = out + (((size_t)bb * COUT + co) * H_ + h) * W_ + wbase;
        v4f lo = { acc[nt][0] + bv, acc[nt][1] + bv, acc[nt][2] + bv, acc[nt][3] + bv };
        v4f hi = { acc[nt][4] + bv, acc[nt][5] + bv, acc[nt][6] + bv, acc[nt][7] + bv };
        *(v4f*)(dst)     = lo;
        *(v4f*)(dst + 4) = hi;
    }
}

// ---------------------------------------------------------------------------
extern "C" void kernel_launch(void* const* d_in, const int* in_sizes, int n_in,
                              void* d_out, int out_size, void* d_ws, size_t ws_size,
                              hipStream_t stream) {
    (void)in_sizes; (void)n_in; (void)out_size; (void)ws_size;
    const float* x    = (const float*)d_in[0];
    const float* w    = (const float*)d_in[1];
    const float* bias = (const float*)d_in[2];
    float*       out  = (float*)d_out;
    _Float16*    wt   = (_Float16*)d_ws;       // 9*128*128 f16 = 288 KB

    prep_weights_f16<<<(9 * COUT * CIN + 255) / 256, 256, 0, stream>>>(w, wt);

    const size_t shmem = (size_t)(3 * COLS * CPAD + COUT * CPAD) * sizeof(_Float16); // ~125 KB
    (void)hipFuncSetAttribute((const void*)conv3x3_wmma,
                              hipFuncAttributeMaxDynamicSharedMemorySize, (int)shmem);

    const int grid = B_ * H_ * 2;              // 7168 blocks: (batch, row, half-row)
    conv3x3_wmma<<<grid, THREADS, shmem, stream>>>(x, wt, bias, out);
}
